// CrossDAttention_8504035246261
// MI455X (gfx1250) — matrix operand order repaired and checked by reading, not compiled, not run on hardware
//
#include <hip/hip_runtime.h>

typedef __attribute__((ext_vector_type(16))) __bf16 v16bf;
typedef __attribute__((ext_vector_type(8)))  __bf16 v8bf;
typedef __attribute__((ext_vector_type(8)))  float  v8f;

namespace {
constexpr int B  = 4, C = 256, H = 32, W = 32;
constexpr int NH = 8, NG = 4, HC = 32, GC = 64;
constexpr int N  = H * W;                       // 1024
constexpr float SCALE = 0.17677669529663687f;   // 32^-0.5
}

__device__ __forceinline__ v8bf ld8(const __bf16* p) {
    return *(const v8bf*)p;
}

// ---------------------------------------------------------------------------
// Tiled GEMM: Out[b] = Wm(MxK) @ X[b](KxN) + bias, bf16 WMMA compute.
// Epilogue modes:
//   0: fp32 Out (B,M,N)
//   1: fp32 Out (B,M,N)  +  bf16 OutT (B,N,M)   (q: fp32 for offset net,
//                                                bf16 token-major for attn)
//   2: bf16 OutT (B,N,M) only                    (k token-major)
//   3: bf16 OutB (B,M,N) only                    (v channel-major)
// ---------------------------------------------------------------------------
__global__ __launch_bounds__(32)
void gemm_wmma_bf16(const float* __restrict__ Wm, const float* __restrict__ bias,
                    const float* __restrict__ X, float* __restrict__ OutF,
                    __bf16* __restrict__ OutB, int mode,
                    int M, int K, int Ncols)
{
    const int lane = threadIdx.x & 31;
    const int col  = lane & 15;
    const int hi   = lane >> 4;
    const int m0   = blockIdx.x * 16;
    const int n0   = blockIdx.y * 16;
    const int b    = blockIdx.z;

    const float* Xb = X + (size_t)b * K * Ncols;

    v8f acc = {};
    for (int k0 = 0; k0 < K; k0 += 32) {
        // A fragment: row m0+col, k = k0 + hi*16 + e
        v16bf a;
        const float* arow = Wm + (size_t)(m0 + col) * K + k0 + hi * 16;
#pragma unroll
        for (int e = 0; e < 8; ++e) {
            a[e]     = (__bf16)arow[e];
            a[e + 8] = (__bf16)arow[8 + e];
        }
        // B fragment: col n0+col, k = k0 + hi*16 + e
        v16bf bf;
        const float* bcol = Xb + (size_t)(k0 + hi * 16) * Ncols + n0 + col;
#pragma unroll
        for (int e = 0; e < 16; ++e) bf[e] = (__bf16)bcol[(size_t)e * Ncols];

        acc = __builtin_amdgcn_wmma_f32_16x16x32_bf16(
                  false, a, false, bf, (short)0, acc, false, false);
    }

    float val[8];
#pragma unroll
    for (int r = 0; r < 8; ++r) val[r] = acc[r] + bias[m0 + r + 8 * hi];

    if (mode <= 1) {
        float* Ob = OutF + (size_t)b * M * Ncols;
#pragma unroll
        for (int r = 0; r < 8; ++r)
            Ob[(size_t)(m0 + r + 8 * hi) * Ncols + n0 + col] = val[r];
    }
    if (mode == 1 || mode == 2) {
        // transposed bf16: (b, n, m) -- contiguous across r, vectorizable
        __bf16* Ot = OutB + ((size_t)b * Ncols + n0 + col) * M + m0 + 8 * hi;
#pragma unroll
        for (int r = 0; r < 8; ++r) Ot[r] = (__bf16)val[r];
    }
    if (mode == 3) {
        __bf16* Ob = OutB + (size_t)b * M * Ncols;
#pragma unroll
        for (int r = 0; r < 8; ++r)
            Ob[(size_t)(m0 + r + 8 * hi) * Ncols + n0 + col] = (__bf16)val[r];
    }
}

// ---------------------------------------------------------------------------
// Offset network: block of 64 threads (thread == channel) per (bg, pixel).
// depthwise 3x3 -> LayerNorm(ch) -> exact GELU -> 1x1 -> tanh -> + ref.
// ---------------------------------------------------------------------------
__global__ __launch_bounds__(64)
void offset_net(const float* __restrict__ q,   const float* __restrict__ dww,
                const float* __restrict__ dwb, const float* __restrict__ lnw,
                const float* __restrict__ lnb, const float* __restrict__ pww,
                float* __restrict__ pos)
{
    __shared__ float sbuf[64];
    const int c  = threadIdx.x;
    const int p  = blockIdx.x;
    const int iy = p >> 5, ix = p & 31;
    const int bg = blockIdx.y;
    const int b  = bg >> 2, g = bg & 3;

    auto blockSum = [&](float v) -> float {
        sbuf[c] = v; __syncthreads();
        for (int s = 32; s > 0; s >>= 1) {
            if (c < s) sbuf[c] += sbuf[c + s];
            __syncthreads();
        }
        float r = sbuf[0]; __syncthreads();
        return r;
    };

    const float* qc = q + ((size_t)b * C + g * GC + c) * N;
    float acc = dwb[c];
#pragma unroll
    for (int dy = 0; dy < 3; ++dy) {
        int yy = iy + dy - 1;
        if (yy < 0 || yy >= H) continue;
#pragma unroll
        for (int dx = 0; dx < 3; ++dx) {
            int xx = ix + dx - 1;
            if (xx < 0 || xx >= W) continue;
            acc += qc[yy * W + xx] * dww[c * 9 + dy * 3 + dx];
        }
    }
    float mu  = blockSum(acc) * (1.f / 64.f);
    float d   = acc - mu;
    float var = blockSum(d * d) * (1.f / 64.f);
    float xn  = d * rsqrtf(var + 1e-5f) * lnw[c] + lnb[c];
    float ge  = 0.5f * xn * (1.f + erff(xn * 0.70710678118f));
    float oy  = blockSum(pww[c]      * ge);
    float ox  = blockSum(pww[GC + c] * ge);
    if (c == 0) {
        float ry = ((0.5f + (float)iy) / 31.f) * 2.f - 1.f;
        float rx = ((0.5f + (float)ix) / 31.f) * 2.f - 1.f;
        pos[((size_t)bg * N + p) * 2 + 0] = tanhf(oy) * (4.f / 31.f) + ry;
        pos[((size_t)bg * N + p) * 2 + 1] = tanhf(ox) * (4.f / 31.f) + rx;
    }
}

// ---------------------------------------------------------------------------
// Deformable bilinear gather of kv features at pos (zeros padding,
// align_corners=True). One thread per (bg, c, p) output element.
// ---------------------------------------------------------------------------
__global__ __launch_bounds__(256)
void deform_sample(const float* __restrict__ kv, const float* __restrict__ pos,
                   float* __restrict__ xs)
{
    const int t  = blockIdx.x * 256 + threadIdx.x;
    const int p  = t & (N - 1);
    const int c  = (t >> 10) & (GC - 1);
    const int bg = t >> 16;

    const float gy = pos[((size_t)bg * N + p) * 2 + 0];
    const float gx = pos[((size_t)bg * N + p) * 2 + 1];
    const float x = (gx + 1.f) * 15.5f;   // (W-1)/2
    const float y = (gy + 1.f) * 15.5f;
    const float x0 = floorf(x), y0 = floorf(y);
    const float wx1 = x - x0, wy1 = y - y0;
    const int ix0 = (int)x0, iy0 = (int)y0;

    const float* img = kv + ((size_t)bg * GC + c) * N;
    float acc = 0.f;
#pragma unroll
    for (int cy = 0; cy < 2; ++cy)
#pragma unroll
        for (int cx = 0; cx < 2; ++cx) {
            int xi = ix0 + cx, yi = iy0 + cy;
            float w = (cx ? wx1 : 1.f - wx1) * (cy ? wy1 : 1.f - wy1);
            if (xi >= 0 && xi < W && yi >= 0 && yi < H)
                acc += img[yi * W + xi] * w;
        }
    xs[(size_t)bg * GC * N + (size_t)c * N + p] = acc;
}

// ---------------------------------------------------------------------------
// Fused flash attention with rpe bias. Block = 128 threads (4 waves); each
// wave owns 16 query rows; keys streamed in chunks of 32. q/k/v are bf16 in
// WMMA-friendly layouts, so every operand feed is a b128 load; the only cvts
// left in the loop are the packed P->bf16 stores to LDS.
// ---------------------------------------------------------------------------
__device__ __forceinline__ float rpe_sample(const float* __restrict__ t,
                                            float gx, float gy)
{
    float x = (gx + 1.f) * 31.f;
    float y = (gy + 1.f) * 31.f;
    float x0 = floorf(x), y0 = floorf(y);
    float wx1 = x - x0, wy1 = y - y0;
    int ix0 = (int)x0, iy0 = (int)y0;
    float acc = 0.f;
#pragma unroll
    for (int cy = 0; cy < 2; ++cy)
#pragma unroll
        for (int cx = 0; cx < 2; ++cx) {
            int xi = ix0 + cx, yi = iy0 + cy;
            float w = (cx ? wx1 : 1.f - wx1) * (cy ? wy1 : 1.f - wy1);
            if (xi >= 0 && xi < 63 && yi >= 0 && yi < 63)
                acc += t[yi * 63 + xi] * w;
        }
    return acc;
}

__global__ __launch_bounds__(128)
void flash_attn(const __bf16* __restrict__ qt,  // (B, N, C) token-major
                const __bf16* __restrict__ kt,  // (B, N, C) token-major
                const __bf16* __restrict__ vb,  // (B, C, N) channel-major
                const float* __restrict__ pos, const float* __restrict__ rpe,
                float* __restrict__ out)
{
    __shared__ float  srpe[63 * 63];
    __shared__ __bf16 sP[4][16 * 32];

    const int tid  = threadIdx.x;
    const int wave = tid >> 5, lane = tid & 31;
    const int col  = lane & 15, hi = lane >> 4;
    const int h = blockIdx.y, b = blockIdx.z;
    const int bg = b * NG + (h >> 1);
    const int m_base = blockIdx.x * 64 + wave * 16;

    const float* rh = rpe + (size_t)h * 63 * 63;
    for (int i = tid; i < 63 * 63; i += 128) srpe[i] = rh[i];
    __syncthreads();

    // Q A-fragment: row = m_base+col, element e -> ch hi*16 + e
    v16bf aq;
    {
        const __bf16* qrow = qt + ((size_t)(b * N + m_base + col)) * C + h * HC;
        v8bf lo = ld8(qrow + hi * 16);
        v8bf hp = ld8(qrow + hi * 16 + 8);
#pragma unroll
        for (int e = 0; e < 8; ++e) { aq[e] = lo[e]; aq[e + 8] = hp[e]; }
    }
    float qgy[8], qgx[8];
#pragma unroll
    for (int r = 0; r < 8; ++r) {
        int m = m_base + r + 8 * hi;
        qgy[r] = ((float)(m >> 5) / 31.f) * 2.f - 1.f;
        qgx[r] = ((float)(m & 31) / 31.f) * 2.f - 1.f;
    }

    const __bf16* ktb = kt + (size_t)b * N * C + h * HC;
    const __bf16* vbb = vb + ((size_t)b * C + h * HC) * N;

    float rowmax[8], rowsum[8];
#pragma unroll
    for (int r = 0; r < 8; ++r) { rowmax[r] = -1e30f; rowsum[r] = 0.f; }
    v8f o0 = {}, o1 = {};

    for (int n0 = 0; n0 < N; n0 += 32) {
        __builtin_prefetch(ktb + (size_t)(n0 + 32) * C, 0, 1);
        // K B-fragments: col = key, element e -> ch = hi*16 + e (contiguous)
        v16bf kf0, kf1;
        {
            const __bf16* k0 = ktb + (size_t)(n0 + col) * C + hi * 16;
            const __bf16* k1 = ktb + (size_t)(n0 + 16 + col) * C + hi * 16;
            v8bf a0 = ld8(k0), a1 = ld8(k0 + 8);
            v8bf b0 = ld8(k1), b1 = ld8(k1 + 8);
#pragma unroll
            for (int e = 0; e < 8; ++e) {
                kf0[e] = a0[e]; kf0[e + 8] = a1[e];
                kf1[e] = b0[e]; kf1[e + 8] = b1[e];
            }
        }
        v8f s0 = {}, s1 = {};
        s0 = __builtin_amdgcn_wmma_f32_16x16x32_bf16(false, aq, false, kf0,
                                                     (short)0, s0, false, false);
        s1 = __builtin_amdgcn_wmma_f32_16x16x32_bf16(false, aq, false, kf1,
                                                     (short)0, s1, false, false);
        // scale + rpe bias (this lane's keys: n0+col, n0+16+col)
#pragma unroll
        for (int half = 0; half < 2; ++half) {
            int key = n0 + half * 16 + col;
            float py = pos[((size_t)bg * N + key) * 2 + 0];
            float px = pos[((size_t)bg * N + key) * 2 + 1];
#pragma unroll
            for (int r = 0; r < 8; ++r) {
                float dy = (qgy[r] - py) * 0.5f;
                float dx = (qgx[r] - px) * 0.5f;
                float bv = rpe_sample(srpe, dx, dy);
                if (half == 0) s0[r] = s0[r] * SCALE + bv;
                else           s1[r] = s1[r] * SCALE + bv;
            }
        }
        // online softmax: row reductions across 16-lane column group
        float nm[8];
#pragma unroll
        for (int r = 0; r < 8; ++r) nm[r] = fmaxf(s0[r], s1[r]);
#pragma unroll
        for (int msk = 1; msk <= 8; msk <<= 1)
#pragma unroll
            for (int r = 0; r < 8; ++r)
                nm[r] = fmaxf(nm[r], __shfl_xor(nm[r], msk, 32));
        float corr[8];
#pragma unroll
        for (int r = 0; r < 8; ++r) {
            float newm = fmaxf(rowmax[r], nm[r]);
            corr[r] = __expf(rowmax[r] - newm);
            rowmax[r] = newm;
            s0[r] = __expf(s0[r] - newm);
            s1[r] = __expf(s1[r] - newm);
        }
        float rs[8];
#pragma unroll
        for (int r = 0; r < 8; ++r) rs[r] = s0[r] + s1[r];
#pragma unroll
        for (int msk = 1; msk <= 8; msk <<= 1)
#pragma unroll
            for (int r = 0; r < 8; ++r) rs[r] += __shfl_xor(rs[r], msk, 32);
#pragma unroll
        for (int r = 0; r < 8; ++r) {
            rowsum[r] = rowsum[r] * corr[r] + rs[r];
            o0[r] *= corr[r];
            o1[r] *= corr[r];
        }
        // P: C-layout -> bf16 A-layout via LDS
        __bf16* myP = sP[wave];
#pragma unroll
        for (int r = 0; r < 8; ++r) {
            myP[(r + 8 * hi) * 32 + col]      = (__bf16)s0[r];
            myP[(r + 8 * hi) * 32 + col + 16] = (__bf16)s1[r];
        }
        __syncthreads();
        v16bf ap;
        {
            v8bf lo = ld8(&myP[col * 32 + hi * 16]);
            v8bf hp = ld8(&myP[col * 32 + hi * 16 + 8]);
#pragma unroll
            for (int e = 0; e < 8; ++e) { ap[e] = lo[e]; ap[e + 8] = hp[e]; }
        }
        __syncthreads();
        // V^T B-fragments: col = channel, element e -> key n0 + hi*16 + e
        v16bf vf0, vf1;
        {
            const __bf16* v0 = vbb + (size_t)col * N + n0 + hi * 16;
            const __bf16* v1 = vbb + (size_t)(col + 16) * N + n0 + hi * 16;
            v8bf a0 = ld8(v0), a1 = ld8(v0 + 8);
            v8bf b0 = ld8(v1), b1 = ld8(v1 + 8);
#pragma unroll
            for (int e = 0; e < 8; ++e) {
                vf0[e] = a0[e]; vf0[e + 8] = a1[e];
                vf1[e] = b0[e]; vf1[e + 8] = b1[e];
            }
        }
        o0 = __builtin_amdgcn_wmma_f32_16x16x32_bf16(false, ap, false, vf0,
                                                     (short)0, o0, false, false);
        o1 = __builtin_amdgcn_wmma_f32_16x16x32_bf16(false, ap, false, vf1,
                                                     (short)0, o1, false, false);
    }
    // normalize and write: out layout (b, h*32+ch, m), fp32
    float* ob = out + ((size_t)b * C + h * HC) * N;
#pragma unroll
    for (int r = 0; r < 8; ++r) {
        float inv = 1.f / rowsum[r];
        int m = m_base + r + 8 * hi;
        ob[(size_t)col        * N + m] = o0[r] * inv;
        ob[(size_t)(col + 16) * N + m] = o1[r] * inv;
    }
}

// ---------------------------------------------------------------------------
extern "C" void kernel_launch(void* const* d_in, const int* in_sizes, int n_in,
                              void* d_out, int out_size, void* d_ws, size_t ws_size,
                              hipStream_t stream)
{
    (void)in_sizes; (void)n_in; (void)out_size; (void)ws_size;
    const float* q_feat  = (const float*)d_in[0];
    const float* kv_feat = (const float*)d_in[1];
    const float* Wq  = (const float*)d_in[2];
    const float* bq  = (const float*)d_in[3];
    const float* Wk  = (const float*)d_in[4];
    const float* bk  = (const float*)d_in[5];
    const float* Wv  = (const float*)d_in[6];
    const float* bv  = (const float*)d_in[7];
    const float* Wo  = (const float*)d_in[8];
    const float* bo  = (const float*)d_in[9];
    const float* dww = (const float*)d_in[10];
    const float* dwb = (const float*)d_in[11];
    const float* lnw = (const float*)d_in[12];
    const float* lnb = (const float*)d_in[13];
    const float* pww = (const float*)d_in[14];
    const float* rpe = (const float*)d_in[15];

    char* wsb = (char*)d_ws;
    float*  qf  = (float*)wsb;   wsb += (size_t)B * C * N * 4;  // fp32 q (offset net)
    float*  xs  = (float*)wsb;   wsb += (size_t)B * C * N * 4;  // sampled kv
    float*  ao  = (float*)wsb;   wsb += (size_t)B * C * N * 4;  // attn out pre-Wo
    float*  pos = (float*)wsb;   wsb += (size_t)B * NG * N * 2 * 4;
    __bf16* qt  = (__bf16*)wsb;  wsb += (size_t)B * N * C * 2;  // (B,N,C)
    __bf16* kt  = (__bf16*)wsb;  wsb += (size_t)B * N * C * 2;  // (B,N,C)
    __bf16* vbf = (__bf16*)wsb;  wsb += (size_t)B * C * N * 2;  // (B,C,N)

    const dim3 gg(C / 16, N / 16, B);   // 16 x 64 x 4 tiles, 1 wave each

    gemm_wmma_bf16<<<gg, 32, 0, stream>>>(Wq, bq, q_feat, qf, qt, 1, C, C, N);
    offset_net<<<dim3(N, B * NG), GC, 0, stream>>>(qf, dww, dwb, lnw, lnb, pww, pos);
    deform_sample<<<(B * C * N) / 256, 256, 0, stream>>>(kv_feat, pos, xs);
    gemm_wmma_bf16<<<gg, 32, 0, stream>>>(Wk, bk, xs, nullptr, kt, 2, C, C, N);
    gemm_wmma_bf16<<<gg, 32, 0, stream>>>(Wv, bv, xs, nullptr, vbf, 3, C, C, N);
    flash_attn<<<dim3(N / 64, NH, B), 128, 0, stream>>>(qt, kt, vbf, pos, rpe, ao);
    gemm_wmma_bf16<<<gg, 32, 0, stream>>>(Wo, bo, ao, (float*)d_out, nullptr, 0, C, C, N);
}
